// Attention_88424786690750
// MI455X (gfx1250) — matrix operand order repaired and checked by reading, not compiled, not run on hardware
//
#include <hip/hip_runtime.h>
#include <hip/hip_bf16.h>

// ---------------------------------------------------------------------------
// Shapes (fixed by the reference)
// ---------------------------------------------------------------------------
#define HIDDEN   768
#define N_HEADS  12
#define HEAD_DIM 64
#define SEQ      2048
#define BATCH    2
#define MROWS    (BATCH * SEQ)          // 4096
#define QKV_ELEMS (BATCH * N_HEADS * SEQ * HEAD_DIM) // 3,145,728

typedef __attribute__((ext_vector_type(16))) __bf16 v16bf;
typedef __attribute__((ext_vector_type(8)))  __bf16 v8bf;
typedef __attribute__((ext_vector_type(8)))  float  v8f;
typedef __attribute__((ext_vector_type(4)))  unsigned v4u;
typedef __attribute__((ext_vector_type(8)))  unsigned v8u;

// ---------------------------------------------------------------------------
// WMMA helper: D = A(16x32 bf16) * B(32x16 bf16) + C(16x16 f32)
// ---------------------------------------------------------------------------
__device__ inline v8f wmma_bf16(v16bf a, v16bf b, v8f c) {
    return __builtin_amdgcn_wmma_f32_16x16x32_bf16(
        /*neg_a=*/false, a, /*neg_b=*/false, b,
        /*c_mod=*/(short)0, c, /*reuse_a=*/false, /*reuse_b=*/false);
}

// A-frag (16x32, MxK) from row-major src. Lane L holds row M=L&15.
// K pattern: lanes<16 -> K 0..15; lanes>=16 -> K 16..31 (matches B-frag loaders).
__device__ inline v16bf load_a_frag(const __bf16* __restrict__ src, int ld,
                                    int mBase, int k0, int lane) {
    const int m    = mBase + (lane & 15);
    const int koff = (lane >> 4) ? 16 : 0;
    const __bf16* p = src + (size_t)m * ld + k0 + koff;
    v8bf lo = *(const v8bf*)(p);        // K = koff .. koff+7
    v8bf hi = *(const v8bf*)(p + 8);    // K = koff+8 .. koff+15
    v16bf a;
#pragma unroll
    for (int i = 0; i < 8; ++i) { a[i] = lo[i]; a[8 + i] = hi[i]; }
    return a;
}

// B-frag (32x16, KxN) where column n is a contiguous row of src (src is [N,K]
// row-major, i.e. B = src^T). Lanes<16 hold K 0..15, lanes>=16 hold K 16..31.
__device__ inline v16bf load_b_frag_rows(const __bf16* __restrict__ src, int ld,
                                         int nBase, int k0, int lane) {
    const int n  = nBase + (lane & 15);
    const int kl = (lane >> 4) ? 16 : 0;
    const __bf16* p = src + (size_t)n * ld + k0 + kl;
    return *(const v16bf*)p;            // 16 contiguous bf16 (32B)
}

// B-frag (32x16, KxN) from a row-major [32,64] tile held in LDS.
__device__ inline v16bf load_b_frag_lds(const __bf16* src, int nBase, int lane) {
    const int n  = nBase + (lane & 15);
    const int kl = (lane >> 4) ? 16 : 0;
    v16bf b;
#pragma unroll
    for (int i = 0; i < 16; ++i) b[i] = src[(kl + i) * HEAD_DIM + n];
    return b;
}

// ---------------------------------------------------------------------------
// TDM: async DMA of a 2D bf16 tile (tile_d1 rows x tile_d0 elems) from global
// memory into LDS.  D# layout per CDNA5 ISA.
// Group0: [1:0]=count=1, [63:32]=lds_addr, [120:64]=global_addr, [127:126]=2.
// Group1: [17:16]=data_size(1->2B), [79:48]=tensor_dim0, [111:80]=tensor_dim1,
//         [127:112]=tile_dim0, [143:128]=tile_dim1, [207:160]=dim0_stride.
// Tracked by TENSORcnt.  Inline asm (portable across toolchains).
// ---------------------------------------------------------------------------
__device__ inline void tdm_load_2d_bf16(unsigned lds_off, const void* gptr,
                                        unsigned tensor_d0, unsigned tensor_d1,
                                        unsigned tile_d0, unsigned tile_d1,
                                        unsigned stride0) {
    const unsigned long long ga = (unsigned long long)gptr;
    v4u g0;
    g0[0] = 1u;                                   // count=1, user-mode
    g0[1] = lds_off;                              // lds_addr (bytes)
    g0[2] = (unsigned)(ga & 0xffffffffu);         // global_addr[31:0]
    g0[3] = (unsigned)((ga >> 32) & 0x01ffffffu)  // global_addr[56:32]
            | (2u << 30);                         // type = 2 ("image")
    v8u g1;
    g1[0] = 1u << 16;                             // data_size = 1 -> 2 bytes
    g1[1] = (tensor_d0 & 0xffffu) << 16;          // tensor_dim0[15:0] @bit48
    g1[2] = ((tensor_d0 >> 16) & 0xffffu)         // tensor_dim0[31:16]
            | ((tensor_d1 & 0xffffu) << 16);      // tensor_dim1[15:0] @bit80
    g1[3] = ((tensor_d1 >> 16) & 0xffffu)         // tensor_dim1[31:16]
            | ((tile_d0 & 0xffffu) << 16);        // tile_dim0 @bit112
    g1[4] = tile_d1 & 0xffffu;                    // tile_dim1 @bit128 (tile_dim2=0)
    g1[5] = stride0;                              // tensor_dim0_stride[31:0] @bit160
    g1[6] = 0;                                    // stride0[47:32], dim1_stride lo
    g1[7] = 0;
    asm volatile("tensor_load_to_lds %0, %1" :: "s"(g0), "s"(g1) : "memory");
}

// ---------------------------------------------------------------------------
// fp32 -> bf16 conversion
// ---------------------------------------------------------------------------
__global__ void cvt_f32_bf16_kernel(const float* __restrict__ in,
                                    __bf16* __restrict__ out, int n) {
    int i = blockIdx.x * blockDim.x + threadIdx.x;
    if (i < n) out[i] = (__bf16)in[i];
}

// ---------------------------------------------------------------------------
// GEMM: C[M,N] = A[M,K](bf16) * W[N,K]^T(bf16), fp32 accumulate.
// MODE 0: write fp32 to [B,H,S,D]   (Q/K pre-RoPE)
// MODE 1: write bf16 to [B,H,S,D]   (V)
// MODE 2: write fp32 row-major [M,N] (final output)
// Block = 256 threads (8 waves); wave tile = 16(M) x 64(N).
// ---------------------------------------------------------------------------
template <int MODE>
__global__ void gemm_bf16_kernel(const __bf16* __restrict__ A, int lda,
                                 const __bf16* __restrict__ W, int ldw,
                                 float* __restrict__ outF,
                                 __bf16* __restrict__ outB,
                                 int M, int N, int K) {
    const int lane  = threadIdx.x & 31;
    const int wave  = threadIdx.x >> 5;
    const int mBase = (blockIdx.x * 8 + wave) * 16;
    const int nBase = blockIdx.y * 64;

    v8f c[4] = {v8f{}, v8f{}, v8f{}, v8f{}};

    for (int k0 = 0; k0 < K; k0 += 32) {
        v16bf a = load_a_frag(A, lda, mBase, k0, lane);
#pragma unroll
        for (int t = 0; t < 4; ++t) {
            v16bf b = load_b_frag_rows(W, ldw, nBase + t * 16, k0, lane);
            c[t] = wmma_bf16(a, b, c[t]);
        }
    }

    const int n0     = lane & 15;
    const int rowAdd = (lane >> 4) ? 8 : 0;
#pragma unroll
    for (int t = 0; t < 4; ++t) {
#pragma unroll
        for (int e = 0; e < 8; ++e) {
            const int r   = mBase + e + rowAdd;     // row in [M]
            const int col = nBase + t * 16 + n0;    // col in [N]
            const float v = c[t][e];
            if (MODE == 2) {
                outF[(size_t)r * N + col] = v;
            } else {
                // (b*SEQ+s, h*64+d) -> [B,H,S,D]
                const int b_ = r >> 11, s = r & (SEQ - 1);
                const int h  = col >> 6, d = col & 63;
                const size_t idx =
                    (((size_t)(b_ * N_HEADS + h) << 11) + s) * HEAD_DIM + d;
                if (MODE == 0) outF[idx] = v;
                else           outB[idx] = (__bf16)v;
            }
        }
    }
}

// ---------------------------------------------------------------------------
// RoPE on fp32 Q/K [B,H,S,D]; emits bf16.
// q'[d] = q[d]*cos(s*f) + (d<32 ? -q[d+32] : q[d-32])*sin(s*f), f = base^-(i/32)
// ---------------------------------------------------------------------------
__global__ void rope_kernel(const float* __restrict__ qf,
                            const float* __restrict__ kf,
                            __bf16* __restrict__ qo,
                            __bf16* __restrict__ ko, int total) {
    int i = blockIdx.x * blockDim.x + threadIdx.x;
    if (i >= total) return;
    const int d  = i & 63;
    const int s  = (i >> 6) & (SEQ - 1);
    const int fi = d & 31;
    const float invf  = __powf(10000.0f, -(float)fi * (1.0f / 32.0f));
    const float ang   = (float)s * invf;
    const float c     = __cosf(ang);
    const float sn    = __sinf(ang);
    {
        const float x = qf[i];
        const float o = (d < 32) ? -qf[i + 32] : qf[i - 32];
        qo[i] = (__bf16)(x * c + o * sn);
    }
    {
        const float x = kf[i];
        const float o = (d < 32) ? -kf[i + 32] : kf[i - 32];
        ko[i] = (__bf16)(x * c + o * sn);
    }
}

// ---------------------------------------------------------------------------
// Flash attention: 1 wave per (b, h, 16-query tile). Causal mask analytic.
// Scores: 2x wmma (D=64). Online softmax in fp32 via 16-lane shuffles.
// V blocks (32x64 bf16) are DMA'd into a double-buffered LDS tile by the
// Tensor Data Mover (TENSORcnt-tracked), overlapping the score WMMAs/softmax.
// P staged through LDS (bf16) to rebuild A-frag layout for P*V (4x wmma).
// Output -> [B, S, H*D] bf16 (row-major for the final GEMM).
// ---------------------------------------------------------------------------
__global__ void __launch_bounds__(32)
attn_kernel(const __bf16* __restrict__ Q, const __bf16* __restrict__ K,
            const __bf16* __restrict__ V, __bf16* __restrict__ O) {
    __shared__ __bf16 pTile[16][32];
    __shared__ __bf16 vbuf[2][32 * HEAD_DIM];   // double-buffered V tile (TDM dest)

    const int lane = threadIdx.x;
    const int tile = blockIdx.x;              // b*H*(S/16) tiles
    const int qt   = tile & (SEQ / 16 - 1);   // query tile 0..127
    const int bh   = tile >> 7;               // b*N_HEADS + h
    const int h    = bh % N_HEADS;
    const int b    = bh / N_HEADS;

    const __bf16* Qb = Q + (size_t)bh * SEQ * HEAD_DIM;
    const __bf16* Kb = K + (size_t)bh * SEQ * HEAD_DIM;
    const __bf16* Vb = V + (size_t)bh * SEQ * HEAD_DIM;

    const int qBase  = qt * 16;
    const int n0     = lane & 15;
    const int rowAdd = (lane >> 4) ? 8 : 0;
    const float scale = 0.125f;               // 1/sqrt(64)

    const v16bf qa0 = load_a_frag(Qb, HEAD_DIM, qBase, 0,  lane);
    const v16bf qa1 = load_a_frag(Qb, HEAD_DIM, qBase, 32, lane);

    v8f o[4] = {v8f{}, v8f{}, v8f{}, v8f{}};
    float mrow[8], lrow[8];
#pragma unroll
    for (int e = 0; e < 8; ++e) { mrow[e] = -1e30f; lrow[e] = 0.0f; }

    const int nKeys   = qBase + 16;           // causal horizon
    const int nBlocks = (nKeys + 31) >> 5;    // 32-key blocks

    const unsigned vbufOff[2] = { (unsigned)(size_t)&vbuf[0][0],
                                  (unsigned)(size_t)&vbuf[1][0] };

    // Prologue: kick off the DMA of the first V block.
    tdm_load_2d_bf16(vbufOff[0], Vb, HEAD_DIM, SEQ, HEAD_DIM, 32, HEAD_DIM);

    for (int blk = 0; blk < nBlocks; ++blk) {
        const int kBase = blk * 32;

        // Kick off the DMA of the next V block (double buffer) and prefetch
        // the next K block while we compute on the current one.
        if (blk + 1 < nBlocks) {
            tdm_load_2d_bf16(vbufOff[(blk + 1) & 1],
                             Vb + (size_t)(kBase + 32) * HEAD_DIM,
                             HEAD_DIM, SEQ, HEAD_DIM, 32, HEAD_DIM);
            __builtin_prefetch(Kb + (size_t)(kBase + 32) * HEAD_DIM, 0, 0);
        }

        // ---- scores for two 16-key tiles ----
        v8f sc[2];
#pragma unroll
        for (int t = 0; t < 2; ++t) {
            v16bf kb0 = load_b_frag_rows(Kb, HEAD_DIM, kBase + t * 16, 0,  lane);
            v16bf kb1 = load_b_frag_rows(Kb, HEAD_DIM, kBase + t * 16, 32, lane);
            v8f s = v8f{};
            s = wmma_bf16(qa0, kb0, s);
            s = wmma_bf16(qa1, kb1, s);
            const int keyc = kBase + t * 16 + n0;
#pragma unroll
            for (int e = 0; e < 8; ++e) {
                const int qr = qBase + e + rowAdd;
                s[e] = s[e] * scale + ((keyc > qr) ? -1e9f : 0.0f);
            }
            sc[t] = s;
        }

        // ---- online softmax update over the 32-key block ----
        float mnew[8], corr[8];
#pragma unroll
        for (int e = 0; e < 8; ++e) {
            float v = fmaxf(sc[0][e], sc[1][e]);
            v = fmaxf(v, __shfl_xor(v, 1, 16));
            v = fmaxf(v, __shfl_xor(v, 2, 16));
            v = fmaxf(v, __shfl_xor(v, 4, 16));
            v = fmaxf(v, __shfl_xor(v, 8, 16));
            mnew[e] = fmaxf(mrow[e], v);
            corr[e] = __expf(mrow[e] - mnew[e]);
            mrow[e] = mnew[e];
        }
#pragma unroll
        for (int t = 0; t < 2; ++t) {
#pragma unroll
            for (int e = 0; e < 8; ++e) {
                const float p = __expf(sc[t][e] - mnew[e]);
                sc[t][e] = p;
                pTile[e + rowAdd][t * 16 + n0] = (__bf16)p;
            }
        }
#pragma unroll
        for (int e = 0; e < 8; ++e) {
            float rs = sc[0][e] + sc[1][e];
            rs += __shfl_xor(rs, 1, 16);
            rs += __shfl_xor(rs, 2, 16);
            rs += __shfl_xor(rs, 4, 16);
            rs += __shfl_xor(rs, 8, 16);
            lrow[e] = lrow[e] * corr[e] + rs;
#pragma unroll
            for (int dc = 0; dc < 4; ++dc) o[dc][e] *= corr[e];
        }

        // ---- transpose P through LDS into A-frag layout ----
        __builtin_amdgcn_wave_barrier();
        asm volatile("s_wait_dscnt 0" ::: "memory");

        v16bf pa;
        {
            const int m    = lane & 15;
            const int koff = (lane >> 4) ? 16 : 0;
#pragma unroll
            for (int i = 0; i < 8; ++i) {
                pa[i]     = pTile[m][koff + i];
                pa[8 + i] = pTile[m][koff + 8 + i];
            }
        }
        __builtin_amdgcn_wave_barrier();

        // ---- wait for this block's V DMA (<=1 leaves the next one in flight) ----
        if (blk + 1 < nBlocks) __builtin_amdgcn_s_wait_tensorcnt(1);
        else                   __builtin_amdgcn_s_wait_tensorcnt(0);

        // ---- O += P * V  (V B-frags from the LDS tile) ----
        const __bf16* vt = &vbuf[blk & 1][0];
#pragma unroll
        for (int dc = 0; dc < 4; ++dc) {
            v16bf vbfrag = load_b_frag_lds(vt, dc * 16, lane);
            o[dc] = wmma_bf16(pa, vbfrag, o[dc]);
        }
    }

    // ---- normalize and store to [B, S, H*D] ----
#pragma unroll
    for (int dc = 0; dc < 4; ++dc) {
#pragma unroll
        for (int e = 0; e < 8; ++e) {
            const int srow = qBase + e + rowAdd;
            const int col  = h * HEAD_DIM + dc * 16 + n0;
            O[((size_t)b * SEQ + srow) * HIDDEN + col] =
                (__bf16)(o[dc][e] / lrow[e]);
        }
    }
}

// ---------------------------------------------------------------------------
// Host-side launcher
// ---------------------------------------------------------------------------
extern "C" void kernel_launch(void* const* d_in, const int* in_sizes, int n_in,
                              void* d_out, int out_size, void* d_ws, size_t ws_size,
                              hipStream_t stream) {
    const float* hs = (const float*)d_in[0];
    const float* wq = (const float*)d_in[1];
    const float* wk = (const float*)d_in[2];
    const float* wv = (const float*)d_in[3];
    const float* wo = (const float*)d_in[4];
    // d_in[5] = attn_mask: exactly triu(-1e9), applied analytically in-kernel.
    float* out = (float*)d_out;

    char*  ws  = (char*)d_ws;
    size_t off = 0;
    auto alloc = [&](size_t bytes) -> void* {
        void* p = ws + off;
        off += (bytes + 255) & ~(size_t)255;
        return p;
    };

    __bf16* Xb  = (__bf16*)alloc((size_t)MROWS * HIDDEN * 2);
    __bf16* Wqb = (__bf16*)alloc((size_t)HIDDEN * HIDDEN * 2);
    __bf16* Wkb = (__bf16*)alloc((size_t)HIDDEN * HIDDEN * 2);
    __bf16* Wvb = (__bf16*)alloc((size_t)HIDDEN * HIDDEN * 2);
    __bf16* Wob = (__bf16*)alloc((size_t)HIDDEN * HIDDEN * 2);
    float*  Qf  = (float*) alloc((size_t)QKV_ELEMS * 4);
    float*  Kf  = (float*) alloc((size_t)QKV_ELEMS * 4);
    __bf16* Vb  = (__bf16*)alloc((size_t)QKV_ELEMS * 2);
    __bf16* Qb  = (__bf16*)alloc((size_t)QKV_ELEMS * 2);
    __bf16* Kb  = (__bf16*)alloc((size_t)QKV_ELEMS * 2);
    __bf16* Ab  = (__bf16*)alloc((size_t)MROWS * HIDDEN * 2);

    // 1) convert inputs to bf16
    {
        const int nX = MROWS * HIDDEN;
        cvt_f32_bf16_kernel<<<(nX + 255) / 256, 256, 0, stream>>>(hs, Xb, nX);
        const int nW = HIDDEN * HIDDEN;
        cvt_f32_bf16_kernel<<<(nW + 255) / 256, 256, 0, stream>>>(wq, Wqb, nW);
        cvt_f32_bf16_kernel<<<(nW + 255) / 256, 256, 0, stream>>>(wk, Wkb, nW);
        cvt_f32_bf16_kernel<<<(nW + 255) / 256, 256, 0, stream>>>(wv, Wvb, nW);
        cvt_f32_bf16_kernel<<<(nW + 255) / 256, 256, 0, stream>>>(wo, Wob, nW);
    }

    // 2) QKV projections (WMMA)
    dim3 ggrid(MROWS / 128, HIDDEN / 64);   // (32, 12)
    gemm_bf16_kernel<0><<<ggrid, 256, 0, stream>>>(Xb, HIDDEN, Wqb, HIDDEN,
                                                   Qf, nullptr, MROWS, HIDDEN, HIDDEN);
    gemm_bf16_kernel<0><<<ggrid, 256, 0, stream>>>(Xb, HIDDEN, Wkb, HIDDEN,
                                                   Kf, nullptr, MROWS, HIDDEN, HIDDEN);
    gemm_bf16_kernel<1><<<ggrid, 256, 0, stream>>>(Xb, HIDDEN, Wvb, HIDDEN,
                                                   nullptr, Vb, MROWS, HIDDEN, HIDDEN);

    // 3) RoPE (fp32) -> bf16 Q/K
    {
        const int nQ = QKV_ELEMS;
        rope_kernel<<<(nQ + 255) / 256, 256, 0, stream>>>(Qf, Kf, Qb, Kb, nQ);
    }

    // 4) causal flash attention (WMMA + TDM double-buffered V)
    attn_kernel<<<BATCH * N_HEADS * (SEQ / 16), 32, 0, stream>>>(Qb, Kb, Vb, Ab);

    // 5) output projection (WMMA) -> fp32 d_out
    gemm_bf16_kernel<2><<<ggrid, 256, 0, stream>>>(Ab, HIDDEN, Wob, HIDDEN,
                                                   out, nullptr, MROWS, HIDDEN, HIDDEN);
}
